// Block3D_66984309949199
// MI455X (gfx1250) — matrix operand order repaired and checked by reading, not compiled, not run on hardware
//
#include <hip/hip_runtime.h>
#include <math.h>

// ---------------------------------------------------------------------------
// Types for CDNA5 WMMA (wave32): V_WMMA_F32_16X16X32_BF16
// ---------------------------------------------------------------------------
typedef __attribute__((ext_vector_type(16))) __bf16 v16bf;
typedef __attribute__((ext_vector_type(8)))  float  v8f;

#define SPATIAL 13824      // 24*24*24
#define SDIM 24
#define PLANE 576          // 24*24
#define CCH 768
#define HIDCH 3072
#define GLUCH 1536
#define EPSV 1e-5f

// LDS tile row stride (elements). 40 bf16 = 80 bytes: every 8-element chunk is
// 16B-aligned (ds_load_b128-able) and rows shift 20 banks => conflict-free.
#define LROW 40

// Probe for the gfx1250 async global->LDS copy builtin (ASYNCcnt path).
#if defined(__has_builtin)
#if __has_builtin(__builtin_amdgcn_global_load_async_to_lds_b128)
#define HAVE_ASYNC_LDS 1
#endif
#endif

#ifdef HAVE_ASYNC_LDS
// Signature discovered from compiler diagnostics: param 0 is
// 'int __attribute__((vector_size(16))) __device__ *'  (AS1 = global).
typedef int v4i_t __attribute__((vector_size(16)));
typedef __attribute__((address_space(1))) v4i_t* as1_v4i;
typedef __attribute__((address_space(3))) v4i_t* as3_v4i;
#endif

// ---------------------------------------------------------------------------
// fp32 -> bf16 bulk convert (for weight matrices), vectorized 4x.
// ---------------------------------------------------------------------------
__global__ __launch_bounds__(256) void f32_to_bf16_kernel(
    const float* __restrict__ in, __bf16* __restrict__ out, long n4)
{
    long i = (long)blockIdx.x * 256 + threadIdx.x;
    const long stride = (long)gridDim.x * 256;
    for (; i < n4; i += stride) {
        const float4 v = *(const float4*)&in[i * 4];
        __bf16* o = &out[i * 4];
        o[0] = (__bf16)v.x; o[1] = (__bf16)v.y;
        o[2] = (__bf16)v.z; o[3] = (__bf16)v.w;
    }
}

// ---------------------------------------------------------------------------
// Per-channel reduction: sum (scaled) and optionally sum of squares.
// ---------------------------------------------------------------------------
__global__ __launch_bounds__(256) void chan_reduce_kernel(
    const float* __restrict__ in, float* __restrict__ osum,
    float* __restrict__ osumsq, float scale)
{
    __shared__ float r1[256];
    __shared__ float r2[256];
    const int c = blockIdx.x;
    const float* p = in + (long)c * SPATIAL;
    float s = 0.f, s2 = 0.f;
    for (int i = threadIdx.x; i < SPATIAL; i += 256) {
        float v = p[i];
        s += v;
        s2 += v * v;
    }
    r1[threadIdx.x] = s;
    r2[threadIdx.x] = s2;
    __syncthreads();
    for (int ofs = 128; ofs > 0; ofs >>= 1) {
        if (threadIdx.x < ofs) {
            r1[threadIdx.x] += r1[threadIdx.x + ofs];
            r2[threadIdx.x] += r2[threadIdx.x + ofs];
        }
        __syncthreads();
    }
    if (threadIdx.x == 0) {
        osum[c] = r1[0] * scale;
        if (osumsq) osumsq[c] = r2[0];
    }
}

// ---------------------------------------------------------------------------
// Conditioning chain: t = relu(y@W1^T+b1)@W2^T+b2 ; tp ; attn ; comb ; kp1
// ---------------------------------------------------------------------------
__global__ __launch_bounds__(768) void condnet1_kernel(
    const float* __restrict__ y,
    const float* __restrict__ lW1, const float* __restrict__ lb1,
    const float* __restrict__ lW2, const float* __restrict__ lb2,
    const float* __restrict__ tpW, const float* __restrict__ tpb,
    const float* __restrict__ aWv, const float* __restrict__ abv,
    const float* __restrict__ aWo, const float* __restrict__ abo,
    const float* __restrict__ knW1, const float* __restrict__ knb1,
    const float* __restrict__ vc,
    float* __restrict__ comb_out, float* __restrict__ kp1_out)
{
    __shared__ float bufA[1536];
    __shared__ float bufB[1536];
    const int t = threadIdx.x;

    bufA[t] = y[t];
    __syncthreads();

    float s = lb1[t];
    for (int j = 0; j < CCH; ++j) s += lW1[t * CCH + j] * bufA[j];
    bufB[t] = fmaxf(s, 0.f);
    __syncthreads();

    s = lb2[t];
    for (int j = 0; j < CCH; ++j) s += lW2[t * CCH + j] * bufB[j];
    bufA[t] = s;
    __syncthreads();

    s = tpb[t];
    for (int j = 0; j < CCH; ++j) s += tpW[t * CCH + j] * bufA[j];
    bufB[t] = s;
    __syncthreads();

    s = abv[t];
    for (int j = 0; j < CCH; ++j) s += aWv[t * CCH + j] * bufB[j];
    bufA[t] = s;
    __syncthreads();

    s = abo[t];
    for (int j = 0; j < CCH; ++j) s += aWo[t * CCH + j] * bufA[j];

    float vcv = vc[t];
    bufB[t] = vcv;
    bufB[CCH + t] = s;
    comb_out[t] = vcv;
    comb_out[CCH + t] = s;
    __syncthreads();

    for (int i = t; i < 2 * CCH * 2; i += 768) {
        float acc = knb1[i];
        for (int j = 0; j < 2 * CCH; ++j) acc += knW1[(long)i * (2 * CCH) + j] * bufB[j];
        kp1_out[i] = fmaxf(acc, 0.f);
    }
}

// ---------------------------------------------------------------------------
// kernels = kn_W2 @ kp1 + kn_b2 ; mod = sigmoid(mod_W @ comb + mod_b)
// ---------------------------------------------------------------------------
__global__ __launch_bounds__(256) void condnet2_kernel(
    const float* __restrict__ comb, const float* __restrict__ kp1,
    const float* __restrict__ knW2, const float* __restrict__ knb2,
    const float* __restrict__ modW, const float* __restrict__ modb,
    float* __restrict__ kern, float* __restrict__ mod)
{
    __shared__ float lk[3072];
    __shared__ float lc[1536];
    for (int i = threadIdx.x; i < 3072; i += 256) lk[i] = kp1[i];
    for (int i = threadIdx.x; i < 1536; i += 256) lc[i] = comb[i];
    __syncthreads();

    const int idx = blockIdx.x * 256 + threadIdx.x;
    if (idx < CCH * 27) {
        float s = knb2[idx];
        for (int j = 0; j < 3072; ++j) s += knW2[(long)idx * 3072 + j] * lk[j];
        kern[idx] = s;
    } else {
        const int m = idx - CCH * 27;
        float s = modb[m];
        for (int j = 0; j < 1536; ++j) s += modW[m * 1536 + j] * lc[j];
        mod[m] = 1.f / (1.f + expf(-s));
    }
}

// ---------------------------------------------------------------------------
// Depthwise 3x3x3 SAME conv with dynamic per-channel kernel + channel scalar
// modulation folded post-conv.
// ---------------------------------------------------------------------------
__global__ __launch_bounds__(256) void dwconv3_mod_kernel(
    const float* __restrict__ in, const float* __restrict__ kern,
    const float* __restrict__ mod, float* __restrict__ out)
{
    __shared__ float pl[3][PLANE];
    __shared__ float kw[27];
    const int c = blockIdx.x / SDIM;
    const int z = blockIdx.x % SDIM;
    const float* inc = in + (long)c * SPATIAL;

    if (threadIdx.x < 27) kw[threadIdx.x] = kern[c * 27 + threadIdx.x];
    for (int i = threadIdx.x; i < 3 * PLANE; i += 256) {
        int dz = i / PLANE, p = i % PLANE;
        int zz = z + dz - 1;
        pl[dz][p] = (zz >= 0 && zz < SDIM) ? inc[zz * PLANE + p] : 0.f;
    }
    __syncthreads();

    const float m = mod[c];
    for (int p = threadIdx.x; p < PLANE; p += 256) {
        const int yy0 = p / SDIM, xx0 = p % SDIM;
        float s = 0.f;
        #pragma unroll
        for (int dz = 0; dz < 3; ++dz) {
            #pragma unroll
            for (int dy = -1; dy <= 1; ++dy) {
                int yy = yy0 + dy;
                if (yy < 0 || yy >= SDIM) continue;
                #pragma unroll
                for (int dx = -1; dx <= 1; ++dx) {
                    int xx = xx0 + dx;
                    if (xx < 0 || xx >= SDIM) continue;
                    s += kw[dz * 9 + (dy + 1) * 3 + (dx + 1)] * pl[dz][yy * SDIM + xx];
                }
            }
        }
        out[(long)c * SPATIAL + z * PLANE + p] = s * m;
    }
}

// ---------------------------------------------------------------------------
// GroupNorm folded into per-channel affine sc/sh for GEMM-1's X loader.
// ---------------------------------------------------------------------------
__global__ __launch_bounds__(768) void gn_coeff_kernel(
    const float* __restrict__ csum, const float* __restrict__ csumsq,
    const float* __restrict__ gn_g, const float* __restrict__ gn_b,
    float* __restrict__ sc, float* __restrict__ sh)
{
    __shared__ float gmu[12], grs[12];
    const int t = threadIdx.x;
    if (t < 12) {
        float S = 0.f, S2 = 0.f;
        for (int c = t * 64; c < t * 64 + 64; ++c) { S += csum[c]; S2 += csumsq[c]; }
        const float n = 64.f * (float)SPATIAL;
        float mu = S / n;
        float var = S2 / n - mu * mu;
        gmu[t] = mu;
        grs[t] = rsqrtf(var + EPSV);
    }
    __syncthreads();
    const int g = t >> 6;
    float scv = grs[g] * gn_g[t];
    sc[t] = scv;
    sh[t] = gn_b[t] - gmu[g] * scv;
}

// ---------------------------------------------------------------------------
// WMMA bf16 GEMM: Out[M,N] = Wb[M,K](bf16) @ X[K,N](fp32->bf16 staged)
//   MODE 0: out = aux * (acc + bias[m]) ; MODE 1: out = acc ;
//   MODE 2: out = aux + acc.  AFFINE: X load applies x*sc[k]+sh[k].
// 128 threads / 4 waves; 64x64 tile; K-step 32.
// W tile: pure bf16 copy -> GLOBAL_LOAD_ASYNC_TO_LDS_B128 when available.
// ---------------------------------------------------------------------------
template <int MODE, bool AFFINE>
__global__ __launch_bounds__(128) void gemm_wmma_kernel(
    const __bf16* __restrict__ Wb, const float* __restrict__ X,
    const float* __restrict__ sc, const float* __restrict__ sh,
    const float* __restrict__ bias, const float* __restrict__ aux,
    float* __restrict__ out, int M, int N, int K)
{
    __shared__ __align__(16) __bf16 Wl[64 * LROW];  // [row][k]
    __shared__ __align__(16) __bf16 Xl[64 * LROW];  // [n][k] transposed

    const int tid  = threadIdx.x;
    const int wave = tid >> 5;
    const int lane = tid & 31;
    const int m0 = blockIdx.y * 64;
    const int n0 = blockIdx.x * 64;

    v8f acc[4] = {};

    const int row   = lane & 15;
    const int abase = (lane < 16) ? 0 : 8;   // A: K[base..base+8) and +16
    const int kb    = (lane < 16) ? 0 : 8;   // B: K[kb..kb+8) and +16

    union Frag { uint4 q[2]; v16bf v; };

    for (int k0 = 0; k0 < K; k0 += 32) {
        // ---- stage W tile (64x32 bf16): 256 x 16B chunks, 2 per thread ----
#ifdef HAVE_ASYNC_LDS
        for (int i = tid; i < 256; i += 128) {
            const int r = i >> 2, c8 = (i & 3) * 8;
            __builtin_amdgcn_global_load_async_to_lds_b128(
                (as1_v4i)(void*)(Wb + (long)(m0 + r) * K + (k0 + c8)),
                (as3_v4i)(void*)(&Wl[r * LROW + c8]),
                0, 0);
        }
#else
        for (int i = tid; i < 256; i += 128) {
            const int r = i >> 2, c8 = (i & 3) * 8;
            *(uint4*)(&Wl[r * LROW + c8]) =
                *(const uint4*)(Wb + (long)(m0 + r) * K + (k0 + c8));
        }
#endif
        // ---- stage X tile (32x64 fp32), transpose + affine + cvt bf16 ----
        for (int i = tid; i < 512; i += 128) {
            const int kk = i >> 4, n4 = (i & 15) * 4;
            const float4 xv = *(const float4*)&X[(long)(k0 + kk) * N + (n0 + n4)];
            float scv = 1.f, shv = 0.f;
            if (AFFINE) { scv = sc[k0 + kk]; shv = sh[k0 + kk]; }
            Xl[(n4 + 0) * LROW + kk] = (__bf16)(xv.x * scv + shv);
            Xl[(n4 + 1) * LROW + kk] = (__bf16)(xv.y * scv + shv);
            Xl[(n4 + 2) * LROW + kk] = (__bf16)(xv.z * scv + shv);
            Xl[(n4 + 3) * LROW + kk] = (__bf16)(xv.w * scv + shv);
        }
        if (k0 + 32 < K) {
            __builtin_prefetch(&Wb[(long)(m0 + (tid >> 1)) * K + (k0 + 32)], 0, 0);
            __builtin_prefetch(&X[(long)(k0 + 32 + (tid >> 6)) * N + n0 + (tid & 63)], 0, 0);
        }
#ifdef HAVE_ASYNC_LDS
        asm volatile("s_wait_asynccnt 0x0" ::: "memory");
#endif
        __syncthreads();

        // ---- A fragment (16x32 bf16): two aligned 16B LDS reads ----
        const __bf16* wr = &Wl[(wave * 16 + row) * LROW];
        Frag fa;
        fa.q[0] = *(const uint4*)(wr + abase);
        fa.q[1] = *(const uint4*)(wr + 16 + abase);
        // ---- 4 B fragments (32x16): two aligned 16B LDS reads ----
        #pragma unroll
        for (int j = 0; j < 4; ++j) {
            const __bf16* xr = &Xl[(j * 16 + row) * LROW + kb];
            Frag fb;
            fb.q[0] = *(const uint4*)(xr);
            fb.q[1] = *(const uint4*)(xr + 16);
            acc[j] = __builtin_amdgcn_wmma_f32_16x16x32_bf16(
                false, fa.v, false, fb.v, (short)0, acc[j], false, false);
        }
        __syncthreads();
    }

    // ---- epilogue ----
    const int mofs = (lane >> 4) << 3;
    #pragma unroll
    for (int j = 0; j < 4; ++j) {
        const int n = n0 + j * 16 + (lane & 15);
        #pragma unroll
        for (int r = 0; r < 8; ++r) {
            const int m = m0 + wave * 16 + r + mofs;
            float v = acc[j][r];
            const long idx = (long)m * N + n;
            if (MODE == 0)      v = aux[idx] * (v + bias[m]);
            else if (MODE == 2) v = aux[idx] + v;
            out[idx] = v;
        }
    }
}

// ---------------------------------------------------------------------------
// Per-position LayerNorm over channels, fused scale/shift.
// ---------------------------------------------------------------------------
__global__ __launch_bounds__(256) void ln_pos_kernel(
    const float* __restrict__ in, const float* __restrict__ w,
    const float* __restrict__ b, float* __restrict__ out, int C)
{
    const int p = blockIdx.x * 256 + threadIdx.x;
    float s = 0.f, s2 = 0.f;
    for (int c = 0; c < C; ++c) {
        float v = in[(long)c * SPATIAL + p];
        s += v; s2 += v * v;
    }
    const float inv = 1.f / (float)C;
    float mu = s * inv;
    float var = s2 * inv - mu * mu;
    float rs = rsqrtf(var + EPSV);
    for (int c = 0; c < C; ++c) {
        float v = in[(long)c * SPATIAL + p];
        out[(long)c * SPATIAL + p] = (v - mu) * rs * w[c] + b[c];
    }
}

// ---------------------------------------------------------------------------
// Fused depthwise 3x3x3 conv (static mlp_dw) + GELU-GLU.
// ---------------------------------------------------------------------------
__global__ __launch_bounds__(256) void dwconv3_glu_kernel(
    const float* __restrict__ h, const float* __restrict__ dw,
    float* __restrict__ g)
{
    __shared__ float pl1[3][PLANE];
    __shared__ float pl2[3][PLANE];
    __shared__ float k1[27], k2[27];
    const int c1 = blockIdx.x / SDIM;
    const int z  = blockIdx.x % SDIM;
    const int c2 = c1 + GLUCH;
    const float* h1 = h + (long)c1 * SPATIAL;
    const float* h2 = h + (long)c2 * SPATIAL;

    if (threadIdx.x < 27) {
        k1[threadIdx.x] = dw[c1 * 27 + threadIdx.x];
        k2[threadIdx.x] = dw[c2 * 27 + threadIdx.x];
    }
    for (int i = threadIdx.x; i < 3 * PLANE; i += 256) {
        int dz = i / PLANE, p = i % PLANE;
        int zz = z + dz - 1;
        bool ok = (zz >= 0 && zz < SDIM);
        pl1[dz][p] = ok ? h1[zz * PLANE + p] : 0.f;
        pl2[dz][p] = ok ? h2[zz * PLANE + p] : 0.f;
    }
    __syncthreads();

    for (int p = threadIdx.x; p < PLANE; p += 256) {
        const int yy0 = p / SDIM, xx0 = p % SDIM;
        float s1 = 0.f, s2 = 0.f;
        #pragma unroll
        for (int dz = 0; dz < 3; ++dz) {
            #pragma unroll
            for (int dy = -1; dy <= 1; ++dy) {
                int yy = yy0 + dy;
                if (yy < 0 || yy >= SDIM) continue;
                #pragma unroll
                for (int dx = -1; dx <= 1; ++dx) {
                    int xx = xx0 + dx;
                    if (xx < 0 || xx >= SDIM) continue;
                    const int kidx = dz * 9 + (dy + 1) * 3 + (dx + 1);
                    const int pidx = yy * SDIM + xx;
                    s1 += k1[kidx] * pl1[dz][pidx];
                    s2 += k2[kidx] * pl2[dz][pidx];
                }
            }
        }
        float gl = 0.5f * s1 * (1.f + erff(s1 * 0.70710678118f));
        g[(long)c1 * SPATIAL + z * PLANE + p] = gl * s2;
    }
}

// ---------------------------------------------------------------------------
// Host-side launch
// ---------------------------------------------------------------------------
extern "C" void kernel_launch(void* const* d_in, const int* in_sizes, int n_in,
                              void* d_out, int out_size, void* d_ws, size_t ws_size,
                              hipStream_t stream)
{
    const float* x       = (const float*)d_in[0];
    const float* y       = (const float*)d_in[1];
    const float* lora_W1 = (const float*)d_in[2];
    const float* lora_b1 = (const float*)d_in[3];
    const float* lora_W2 = (const float*)d_in[4];
    const float* lora_b2 = (const float*)d_in[5];
    const float* tp_W    = (const float*)d_in[6];
    const float* tp_b    = (const float*)d_in[7];
    const float* attn_Wv = (const float*)d_in[8];
    const float* attn_bv = (const float*)d_in[9];
    const float* attn_Wo = (const float*)d_in[10];
    const float* attn_bo = (const float*)d_in[11];
    const float* kn_W1   = (const float*)d_in[12];
    const float* kn_b1   = (const float*)d_in[13];
    const float* kn_W2   = (const float*)d_in[14];
    const float* kn_b2   = (const float*)d_in[15];
    const float* mod_W   = (const float*)d_in[16];
    const float* mod_b   = (const float*)d_in[17];
    const float* gn_g    = (const float*)d_in[18];
    const float* gn_b    = (const float*)d_in[19];
    const float* op_W    = (const float*)d_in[20];
    const float* op_b    = (const float*)d_in[21];
    const float* n2_w    = (const float*)d_in[22];
    const float* n2_b    = (const float*)d_in[23];
    const float* n3_w    = (const float*)d_in[24];
    const float* n3_b    = (const float*)d_in[25];
    const float* mlp_Wi  = (const float*)d_in[26];
    const float* mlp_dw  = (const float*)d_in[27];
    const float* mlp_Wo  = (const float*)d_in[28];

    float* ws = (float*)d_ws;
    const long CV = (long)CCH * SPATIAL;
    const long HV = (long)HIDCH * SPATIAL;
    const long GV = (long)GLUCH * SPATIAL;
    // small scratch
    float* vc    = ws + 0;
    float* comb  = ws + 1024;
    float* kp1   = ws + 3072;
    float* kern  = ws + 8192;
    float* mod   = ws + 30720;
    float* csum  = ws + 31744;
    float* csum2 = ws + 32768;
    float* sc    = ws + 33792;
    float* sh    = ws + 34816;
    // big buffers
    float* obuf = ws + 65536;      // conv out -> later xn -> later x3
    float* x2   = obuf + CV;
    float* hbuf = x2 + CV;
    float* gbuf = hbuf + HV;
    float* xn   = obuf;
    float* x3   = obuf;
    // bf16 weight copies
    __bf16* opWb   = (__bf16*)(gbuf + GV);
    __bf16* mlpWib = opWb + (long)CCH * CCH;
    __bf16* mlpWob = mlpWib + (long)HIDCH * CCH;

    // 0. weight fp32->bf16 copies (independent of data path)
    f32_to_bf16_kernel<<<512, 256, 0, stream>>>(op_W, opWb, (long)CCH * CCH / 4);
    f32_to_bf16_kernel<<<1024, 256, 0, stream>>>(mlp_Wi, mlpWib, (long)HIDCH * CCH / 4);
    f32_to_bf16_kernel<<<1024, 256, 0, stream>>>(mlp_Wo, mlpWob, (long)CCH * GLUCH / 4);

    // 1. vc = mean(x, spatial)
    chan_reduce_kernel<<<CCH, 256, 0, stream>>>(x, vc, nullptr, 1.f / (float)SPATIAL);
    // 2. conditioning GEMV chain
    condnet1_kernel<<<1, 768, 0, stream>>>(y, lora_W1, lora_b1, lora_W2, lora_b2,
                                           tp_W, tp_b, attn_Wv, attn_bv, attn_Wo, attn_bo,
                                           kn_W1, kn_b1, vc, comb, kp1);
    // 3. dynamic kernels + modulation gates
    condnet2_kernel<<<(CCH * 27 + CCH) / 256, 256, 0, stream>>>(
        comb, kp1, kn_W2, kn_b2, mod_W, mod_b, kern, mod);
    // 4. modulated dynamic depthwise conv
    dwconv3_mod_kernel<<<CCH * SDIM, 256, 0, stream>>>(x, kern, mod, obuf);
    // 5-6. groupnorm stats -> per-channel affine
    chan_reduce_kernel<<<CCH, 256, 0, stream>>>(obuf, csum, csum2, 1.f);
    gn_coeff_kernel<<<1, 768, 0, stream>>>(csum, csum2, gn_g, gn_b, sc, sh);
    // 7. x2 = x * (op_W @ gn(obuf) + op_b)
    gemm_wmma_kernel<0, true><<<dim3(SPATIAL / 64, CCH / 64), 128, 0, stream>>>(
        opWb, obuf, sc, sh, op_b, x, x2, CCH, SPATIAL, CCH);
    // 8. xn = LN(x2)
    ln_pos_kernel<<<SPATIAL / 256, 256, 0, stream>>>(x2, n2_w, n2_b, xn, CCH);
    // 9. h = mlp_Wi @ xn
    gemm_wmma_kernel<1, false><<<dim3(SPATIAL / 64, HIDCH / 64), 128, 0, stream>>>(
        mlpWib, xn, nullptr, nullptr, nullptr, nullptr, hbuf, HIDCH, SPATIAL, CCH);
    // 10. g = gelu(dwconv(h_lo)) * dwconv(h_hi)
    dwconv3_glu_kernel<<<GLUCH * SDIM, 256, 0, stream>>>(hbuf, mlp_dw, gbuf);
    // 11. x3 = x2 + mlp_Wo @ g
    gemm_wmma_kernel<2, false><<<dim3(SPATIAL / 64, CCH / 64), 128, 0, stream>>>(
        mlpWob, gbuf, nullptr, nullptr, nullptr, x2, x3, CCH, SPATIAL, GLUCH);
    // 12. out = LN(x3)
    ln_pos_kernel<<<SPATIAL / 256, 256, 0, stream>>>(x3, n3_w, n3_b, (float*)d_out, CCH);
}
